// Model_69672959476023
// MI455X (gfx1250) — matrix operand order repaired and checked by reading, not compiled, not run on hardware
//
#include <hip/hip_runtime.h>
#include <hip/hip_bf16.h>
#include <math.h>

// Problem constants
#define B_DIM 1024
#define L_DIM 320
#define D_DIM 128
#define E_DIM 8
#define P_DIM 96
#define NOISE_EPS 0.01f
#define LOSS_COEF 0.01f

typedef __attribute__((ext_vector_type(16))) __bf16 v16bf;
typedef __attribute__((ext_vector_type(16))) unsigned short v16u;
typedef __attribute__((ext_vector_type(8))) float v8f;

__device__ __forceinline__ unsigned short f2bf(float f) {
  unsigned int u = __builtin_bit_cast(unsigned int, f);
  u += 0x7fffu + ((u >> 16) & 1u);   // round-to-nearest-even
  return (unsigned short)(u >> 16);
}

__device__ __forceinline__ float wave_sum32(float v) {
#pragma unroll
  for (int off = 16; off > 0; off >>= 1) v += __shfl_xor(v, off, 32);
  return v;
}

__device__ __forceinline__ float cdf_f(float v) {
  return 0.5f * (1.0f + erff(v * 0.70710678118654752f));
}

// fragment-layout address helpers (16-bit A 16x32 / B 32x16, wave32)
__device__ __forceinline__ int a_frag_addr(int kl, int p) {
  const int pt = p >> 4;
  const int m  = p & 15;
  const int kh = (kl >> 3) & 1;
  const int j  = (kl & 7) + ((kl >= 16) ? 8 : 0);
  return ((pt * 32 + kh * 16 + m) << 4) | j;
}
__device__ __forceinline__ int b_frag_addr(int kl, int d) {
  const int dt = d >> 4;
  const int n  = d & 15;
  const int kh = (kl >> 3) & 1;
  const int j  = (kl & 7) + ((kl >= 16) ? 8 : 0);
  return ((dt * 32 + kh * 16 + n) << 4) | j;
}

// ---------------------------------------------------------------------------
// zero the 16-float load/importance accumulators in workspace
__global__ void init_acc_kernel(float* acc) {
  if (threadIdx.x < 16) acc[threadIdx.x] = 0.0f;
}

// ---------------------------------------------------------------------------
// Fused gating + top-2 MoE GEMM. One block per batch element b.
// 256 threads = 8 wave32.
__global__ __launch_bounds__(256) void moe_fused_kernel(
    const float* __restrict__ x,        // (B,L,D)
    const float* __restrict__ noise,    // (B,E)
    const float* __restrict__ start_w,  // (D,1)
    const float* __restrict__ start_b,  // (1,)
    const float* __restrict__ w_gate,   // (L,E)
    const float* __restrict__ w_noise,  // (L,E)
    const float* __restrict__ expert_w, // (E,L,P)
    const float* __restrict__ expert_b, // (E,P,D)
    float* __restrict__ out,            // (B,P,D)
    float* __restrict__ load_acc,       // (E,) atomic
    float* __restrict__ imp_acc)        // (E,) atomic
{
  const int b    = blockIdx.x;
  const int t    = threadIdx.x;
  const int lane = t & 31;
  const int wid  = t >> 5;

  __shared__ float glds[L_DIM];          // g[b, :]
  __shared__ float cleanL[E_DIM];
  __shared__ float nstdL[E_DIM];
  __shared__ int   s_e[2];
  __shared__ float s_g[2];
  // chunks stored in WMMA *fragment* order: [tile][lane][16 bf16 contiguous]
  __shared__ __align__(32) unsigned short wfrag[6 * 32 * 16];  // 6 KB
  __shared__ __align__(32) unsigned short xfrag[8 * 32 * 16];  // 8 KB

  const float* xb = x + (size_t)b * (L_DIM * D_DIM);

  // ---- Phase 1: g[l] = x[b,l,:] . start_w + start_b  (one wave per row) ---
  {
    const float4 sw = ((const float4*)start_w)[lane];  // 32 lanes x 4 = 128
    const float  sb = start_b[0];
    for (int l = wid; l < L_DIM; l += 8) {
      const float4 xv = ((const float4*)(xb + l * D_DIM))[lane];
      float p = xv.x * sw.x + xv.y * sw.y + xv.z * sw.z + xv.w * sw.w;
      p = wave_sum32(p);
      if (lane == 0) glds[l] = p + sb;
    }
  }
  __syncthreads();

  // ---- Phase 2: clean logits / noise std per expert (wave e -> expert e) --
  {
    const int e = wid;  // 8 waves, 8 experts
    float cg = 0.0f, cn = 0.0f;
#pragma unroll
    for (int i = 0; i < L_DIM / 32; ++i) {
      const int l = lane + 32 * i;
      const float gv = glds[l];
      cg += gv * w_gate[l * E_DIM + e];
      cn += gv * w_noise[l * E_DIM + e];
    }
    cg = wave_sum32(cg);
    cn = wave_sum32(cn);
    if (lane == 0) {
      cleanL[e] = cg;
      const float sp = (cn > 20.0f) ? cn : log1pf(expf(cn));  // softplus
      nstdL[e] = sp + NOISE_EPS;
    }
  }
  __syncthreads();

  // ---- Phase 2b: serial top-3 / gates / load probs on thread 0 -----------
  if (t == 0) {
    float noisy[E_DIM], cleanc[E_DIM];
#pragma unroll
    for (int e = 0; e < E_DIM; ++e) {
      const float c = cleanL[e];
      noisy[e]  = c + noise[(size_t)b * E_DIM + e] * nstdL[e];
      cleanc[e] = isnan(c) ? 0.0f : c;
    }
    // top-3 (first-index wins ties, like jax top_k)
    int   idx[3];
    float val[3];
    bool taken[E_DIM];
#pragma unroll
    for (int e = 0; e < E_DIM; ++e) taken[e] = false;
#pragma unroll
    for (int r = 0; r < 3; ++r) {
      int   bi = -1;
      float bv = -INFINITY;
      for (int e = 0; e < E_DIM; ++e)
        if (!taken[e] && noisy[e] > bv) { bv = noisy[e]; bi = e; }
      idx[r] = bi; val[r] = bv; taken[bi] = true;
    }
    // softmax over top-2
    const float ex1 = expf(val[1] - val[0]);
    const float den = 1.0f + ex1;
    const float g0 = 1.0f / den;
    const float g1 = ex1 / den;
    s_e[0] = idx[0]; s_e[1] = idx[1];
    s_g[0] = g0;     s_g[1] = g1;
    atomicAdd(&imp_acc[idx[0]], g0);
    atomicAdd(&imp_acc[idx[1]], g1);
    const float thr_in  = val[2];  // (K)-th  (3rd largest)
    const float thr_out = val[1];  // (K-1)th (2nd largest)
#pragma unroll
    for (int e = 0; e < E_DIM; ++e) {
      const float ns = nstdL[e];
      const float p = (noisy[e] > thr_in) ? cdf_f((cleanc[e] - thr_in) / ns)
                                          : cdf_f((cleanc[e] - thr_out) / ns);
      atomicAdd(&load_acc[e], p);
    }
  }
  __syncthreads();

  const int   e0 = s_e[0], e1 = s_e[1];
  const float g0 = s_g[0], g1 = s_g[1];
  const float* w0 = expert_w + (size_t)e0 * (L_DIM * P_DIM);
  const float* w1 = expert_w + (size_t)e1 * (L_DIM * P_DIM);

  // ---- Phase 3: out[b] = (g0*W[e0] + g1*W[e1])^T @ x[b], WMMA bf16 -------
  // Tiles: 6 p-tiles x 8 d-tiles; wave `wid` owns d-tile wid (reuses B frag).
  v8f acc[6];
#pragma unroll
  for (int pt = 0; pt < 6; ++pt) acc[pt] = (v8f)(0.0f);

  for (int kc = 0; kc < L_DIM / 32; ++kc) {
    const int kb = kc * 32;
    // stage gate-combined weight chunk (32 x 96) into fragment layout
    // 3072 elems = 768 float4 pairs, 3 per thread
#pragma unroll
    for (int ii = 0; ii < 3; ++ii) {
      const int i4 = t + ii * 256;          // float4 index
      const int i  = i4 * 4;                // element index: kl*96 + p
      const float4 a4 = *(const float4*)(w0 + kb * P_DIM + i);
      const float4 b4 = *(const float4*)(w1 + kb * P_DIM + i);
      const int kl = i / P_DIM;
      const int p  = i % P_DIM;             // multiple of 4, same row for all 4
      wfrag[a_frag_addr(kl, p + 0)] = f2bf(g0 * a4.x + g1 * b4.x);
      wfrag[a_frag_addr(kl, p + 1)] = f2bf(g0 * a4.y + g1 * b4.y);
      wfrag[a_frag_addr(kl, p + 2)] = f2bf(g0 * a4.z + g1 * b4.z);
      wfrag[a_frag_addr(kl, p + 3)] = f2bf(g0 * a4.w + g1 * b4.w);
    }
    // stage x chunk (32 x 128) into fragment layout: 1024 float4, 4 per thread
#pragma unroll
    for (int ii = 0; ii < 4; ++ii) {
      const int i4 = t + ii * 256;
      const int i  = i4 * 4;                // element index: kl*128 + d
      const float4 x4 = *(const float4*)(xb + (size_t)kb * D_DIM + i);
      const int kl = i >> 7;
      const int d  = i & 127;
      xfrag[b_frag_addr(kl, d + 0)] = f2bf(x4.x);
      xfrag[b_frag_addr(kl, d + 1)] = f2bf(x4.y);
      xfrag[b_frag_addr(kl, d + 2)] = f2bf(x4.z);
      xfrag[b_frag_addr(kl, d + 3)] = f2bf(x4.w);
    }
    __syncthreads();

    // B fragment: one contiguous 32-byte LDS load per lane
    const v16u bsu = *(const v16u*)&xfrag[(wid * 32 + lane) << 4];
    const v16bf bfrag = __builtin_bit_cast(v16bf, bsu);

#pragma unroll
    for (int pt = 0; pt < 6; ++pt) {
      const v16u asu = *(const v16u*)&wfrag[(pt * 32 + lane) << 4];
      const v16bf afrag = __builtin_bit_cast(v16bf, asu);
      acc[pt] = __builtin_amdgcn_wmma_f32_16x16x32_bf16(
          false, afrag, false, bfrag, (short)0, acc[pt], false, false);
    }
    __syncthreads();
  }

  // ---- Phase 4: add gated bias, store ------------------------------------
  {
    const float* b0 = expert_b + (size_t)e0 * (P_DIM * D_DIM);
    const float* b1 = expert_b + (size_t)e1 * (P_DIM * D_DIM);
    float* ob = out + (size_t)b * (P_DIM * D_DIM);
    const int n  = lane & 15;
    const int mh = (lane >> 4) * 8;
    const int d  = wid * 16 + n;
#pragma unroll
    for (int pt = 0; pt < 6; ++pt) {
#pragma unroll
      for (int r = 0; r < 8; ++r) {
        const int p = pt * 16 + r + mh;
        const int o = p * D_DIM + d;
        ob[o] = acc[pt][r] + g0 * b0[o] + g1 * b1[o];
      }
    }
  }
}

// ---------------------------------------------------------------------------
// balance loss from accumulated importance/load; writes the two scalar outs
__global__ void loss_kernel(const float* __restrict__ load_acc,
                            const float* __restrict__ imp_acc,
                            float* __restrict__ out_tail) {
  float si = 0.0f, sl = 0.0f;
#pragma unroll
  for (int e = 0; e < E_DIM; ++e) { si += imp_acc[e]; sl += load_acc[e]; }
  const float mi = si / E_DIM, ml = sl / E_DIM;
  float vi = 0.0f, vl = 0.0f;
#pragma unroll
  for (int e = 0; e < E_DIM; ++e) {
    const float di = imp_acc[e] - mi; vi += di * di;
    const float dl = load_acc[e] - ml; vl += dl * dl;
  }
  vi /= (E_DIM - 1); vl /= (E_DIM - 1);
  const float loss = LOSS_COEF * (vi / (mi * mi + 1e-10f) + vl / (ml * ml + 1e-10f));
  out_tail[0] = loss;
  out_tail[1] = 0.0f;  // regu_sum
}

// ---------------------------------------------------------------------------
extern "C" void kernel_launch(void* const* d_in, const int* in_sizes, int n_in,
                              void* d_out, int out_size, void* d_ws, size_t ws_size,
                              hipStream_t stream) {
  // setup_inputs dict order
  const float* x        = (const float*)d_in[0];
  // d_in[1] = x_mark_enc (unused by reference)
  const float* noise    = (const float*)d_in[2];
  const float* start_w  = (const float*)d_in[3];
  const float* start_b  = (const float*)d_in[4];
  const float* w_gate   = (const float*)d_in[5];
  const float* w_noise  = (const float*)d_in[6];
  const float* expert_w = (const float*)d_in[7];
  const float* expert_b = (const float*)d_in[8];

  float* out = (float*)d_out;
  float* out_tail = out + (size_t)B_DIM * P_DIM * D_DIM;  // [loss, regu_sum]

  float* load_acc = (float*)d_ws;       // E floats
  float* imp_acc  = load_acc + E_DIM;   // E floats

  init_acc_kernel<<<1, 32, 0, stream>>>(load_acc);
  moe_fused_kernel<<<B_DIM, 256, 0, stream>>>(
      x, noise, start_w, start_b, w_gate, w_noise, expert_w, expert_b,
      out, load_acc, imp_acc);
  loss_kernel<<<1, 1, 0, stream>>>(load_acc, imp_acc, out_tail);
}
